// RNNClassifier_31791347925597
// MI455X (gfx1250) — matrix operand order repaired and checked by reading, not compiled, not run on hardware
//
#include <hip/hip_runtime.h>
#include <hip/hip_bf16.h>

#define VOCAB  32000
#define SEQ    512
#define BATCH  128
#define EMB    256
#define HID    512
#define NCLASS 16
#define MROWS  (SEQ * BATCH)

#define RNN_NWG     16
#define RNN_COLS    (HID / RNN_NWG)   // 32 columns of h per persistent workgroup
#define LDSW_STRIDE 520               // 512 + 8 bf16 pad -> 1040B rows, 16B aligned, bank-spread
#define LDSB_STRIDE 264               // 256 + 8 bf16 pad -> 528B rows, 16B aligned, bank-spread

typedef __bf16 bf16;
typedef __attribute__((ext_vector_type(16))) __bf16 v16bf;
typedef __attribute__((ext_vector_type(8)))  __bf16 v8bf;
typedef __attribute__((ext_vector_type(8)))  float  v8f;

// ---------------- WMMA fragment loaders (wave32 layouts per CDNA5 ISA) ----------------
// A-matrix 16x32 bf16: lane = 16*khalf + m.
//   elems 0..7  -> K = kb + khalf*8 + 0..7      (VGPR0..3)
//   elems 8..15 -> K = kb + 16 + khalf*8 + 0..7 (VGPR4..7)
__device__ __forceinline__ v16bf load_fragA_bf16(const bf16* __restrict__ row, int kb, int khalf) {
  v8bf lo = *reinterpret_cast<const v8bf*>(row + kb + khalf * 8);
  v8bf hi = *reinterpret_cast<const v8bf*>(row + kb + 16 + khalf * 8);
  v16bf r;
#pragma unroll
  for (int i = 0; i < 8; ++i) { r[i] = lo[i]; r[i + 8] = hi[i]; }
  return r;
}

// B-matrix 32x16 bf16: lane&15 selects column N; same K order as A:
// elems 0..7 -> K = kb + khalf*8 + 0..7, elems 8..15 -> K = kb + 16 + khalf*8 + 0..7.
__device__ __forceinline__ v16bf load_fragB_bf16(const bf16* __restrict__ colrow, int kb, int khalf) {
  v8bf lo = *reinterpret_cast<const v8bf*>(colrow + kb + khalf * 8);
  v8bf hi = *reinterpret_cast<const v8bf*>(colrow + kb + 16 + khalf * 8);
  v16bf r;
#pragma unroll
  for (int i = 0; i < 8; ++i) { r[i] = lo[i]; r[i + 8] = hi[i]; }
  return r;
}

// ---------------- Prep: one-time f32 -> bf16 of emb and W_ih ----------------
// Removes per-tile v_cvt_pk_bf16_f32 work from the GEMM and halves gather bandwidth.
#define NCH_EMB (VOCAB * EMB / 8)     // 1,024,000 8-element chunks
#define NCH_WIH (HID * EMB / 8)       //    16,384
__global__ void __launch_bounds__(256)
k_prep(const float* __restrict__ emb, const float* __restrict__ W_ih,
       bf16* __restrict__ emb_bf, bf16* __restrict__ wih_bf)
{
  const int i = blockIdx.x * 256 + threadIdx.x;
  if (i >= NCH_EMB + NCH_WIH) return;
  const float* src;
  bf16* dst;
  int j;
  if (i < NCH_EMB) { src = emb;  dst = emb_bf; j = i; }
  else             { src = W_ih; dst = wih_bf; j = i - NCH_EMB; }
  const float4 q0 = *reinterpret_cast<const float4*>(src + (size_t)j * 8);
  const float4 q1 = *reinterpret_cast<const float4*>(src + (size_t)j * 8 + 4);
  v8bf v;
  v[0] = (bf16)q0.x; v[1] = (bf16)q0.y; v[2] = (bf16)q0.z; v[3] = (bf16)q0.w;
  v[4] = (bf16)q1.x; v[5] = (bf16)q1.y; v[6] = (bf16)q1.z; v[7] = (bf16)q1.w;
  *reinterpret_cast<v8bf*>(dst + (size_t)j * 8) = v;
}

// ---------------- Kernel A: fused embedding gather + input projection ----------------
// xproj[i, h] = sum_e emb[x[i], e] * W_ih[h, e] + b_ih[h] + b_hh[h]
// Workgroup(8 waves) -> 128 rows x 64 cols; W_ih bf16 slice staged once in LDS (shared by waves).
__global__ void __launch_bounds__(256)
k_embed_xproj(const int* __restrict__ x, const bf16* __restrict__ emb_bf,
              const bf16* __restrict__ wih_bf, const float* __restrict__ b_ih,
              const float* __restrict__ b_hh, float* __restrict__ xproj)
{
  __shared__ bf16 ldsb[64 * LDSB_STRIDE];   // 64 cols x 256 K, ~33 KiB

  const int lane  = threadIdx.x & 31;
  const int wave  = threadIdx.x >> 5;
  const int l16   = lane & 15;
  const int khalf = lane >> 4;

  const int rowbase = blockIdx.x * 128 + wave * 16;  // row in flattened [SEQ*BATCH]
  const int colbase = blockIdx.y * 64;               // hidden-dim tile

  // Stage B slice: ldsb[c][k] = W_ih[colbase + c][k]  (16B chunks, coalesced)
  for (int i = threadIdx.x; i < 64 * EMB / 8; i += 256) {
    const int c  = i >> 5;            // / (EMB/8)
    const int kc = (i & 31) * 8;
    *reinterpret_cast<v8bf*>(&ldsb[c * LDSB_STRIDE + kc]) =
        *reinterpret_cast<const v8bf*>(wih_bf + (size_t)(colbase + c) * EMB + kc);
  }
  __syncthreads();

  const int erow = x[rowbase + l16];                 // embedding gather index
  const bf16* arow = emb_bf + (size_t)erow * EMB;
  const bf16* brow[4] = { &ldsb[(0  + l16) * LDSB_STRIDE], &ldsb[(16 + l16) * LDSB_STRIDE],
                          &ldsb[(32 + l16) * LDSB_STRIDE], &ldsb[(48 + l16) * LDSB_STRIDE] };

  v8f acc[4] = {};
#pragma unroll
  for (int kb = 0; kb < EMB; kb += 32) {
    const v16bf a = load_fragA_bf16(arow, kb, khalf);
#pragma unroll
    for (int nt = 0; nt < 4; ++nt) {
      const v16bf b = load_fragB_bf16(brow[nt], kb, khalf);
      acc[nt] = __builtin_amdgcn_wmma_f32_16x16x32_bf16(
          false, a, false, b, (short)0, acc[nt], false, false);
    }
  }

  const int m0 = rowbase + khalf * 8;  // C/D layout: row = m0 + vgpr, col = lane&15
#pragma unroll
  for (int nt = 0; nt < 4; ++nt) {
    const int col = colbase + nt * 16 + l16;
    const float bias = b_ih[col] + b_hh[col];
#pragma unroll
    for (int r = 0; r < 8; ++r) {
      xproj[(size_t)(m0 + r) * HID + col] = acc[nt][r] + bias;
    }
  }
}

// ---------------- grid-wide barrier: cluster-HW fast path + atomic fallback ----------------
__device__ __forceinline__ void grid_barrier(unsigned* __restrict__ ctr, unsigned target,
                                             bool use_cluster) {
  __threadfence();                 // release h writes to device scope
  __syncthreads();
  if (use_cluster) {
    // CDNA5 cluster split-barrier (-3): one wave per workgroup signals, all waves wait.
    if (threadIdx.x == 0)
      asm volatile("s_barrier_signal -3" ::: "memory");
    asm volatile("s_barrier_wait -3" ::: "memory");
    __threadfence();               // acquire remote h writes
    __syncthreads();
  } else {
    if (threadIdx.x == 0) {
      __hip_atomic_fetch_add(ctr, 1u, __ATOMIC_ACQ_REL, __HIP_MEMORY_SCOPE_AGENT);
      while (__hip_atomic_load(ctr, __ATOMIC_ACQUIRE, __HIP_MEMORY_SCOPE_AGENT) < target)
        __builtin_amdgcn_s_sleep(1);
    }
    __syncthreads();
  }
}

// ---------------- Kernel B: sequential recurrence h = tanh(xproj_t + h @ W_hh^T) ----------------
// 16 persistent workgroups, each owns 32 columns; W_hh^T slice staged in LDS as bf16.
// h double-buffered in global memory as bf16 (L2 resident, 128 KiB).
__global__ void __launch_bounds__(256)
k_rnn_scan(const float* __restrict__ W_hh, const float* __restrict__ xproj,
           bf16* __restrict__ hb0, bf16* __restrict__ hb1, unsigned* __restrict__ ctr)
{
  __shared__ bf16 ldsw[RNN_COLS * LDSW_STRIDE];   // ~32.5 KiB of 320 KiB

  const int lane  = threadIdx.x & 31;
  const int wave  = threadIdx.x >> 5;
  const int l16   = lane & 15;
  const int khalf = lane >> 4;
  const int colbase = blockIdx.x * RNN_COLS;
  const int rowbase = wave * 16;                   // batch rows owned by this wave

  // If this dispatch was launched as a workgroup-cluster, use the HW cluster barrier.
  unsigned clid;
  asm volatile("s_getreg_b32 %0, hwreg(HW_REG_IB_STS2, 6, 4)" : "=s"(clid));
  const bool use_cluster = (clid != 0);

  // Stage bf16 W_hh slice: ldsw[c][k] = W_hh[colbase + c][k]  (B[k][n] = W_hh[n][k])
  for (int i = threadIdx.x; i < RNN_COLS * HID; i += 256) {
    const int c = i >> 9;            // / HID
    const int k = i & (HID - 1);
    ldsw[c * LDSW_STRIDE + k] = (bf16)W_hh[(size_t)(colbase + c) * HID + k];
  }
  __syncthreads();

  const bf16* b0row = &ldsw[(0  + l16) * LDSW_STRIDE];
  const bf16* b1row = &ldsw[(16 + l16) * LDSW_STRIDE];

  for (int t = 0; t < SEQ; ++t) {
    const bf16* hin  = (t & 1) ? hb1 : hb0;
    bf16*       hout = (t & 1) ? hb0 : hb1;

    // Prefetch next timestep's xproj tile (16 KiB) into cache while WMMAs run.
    if (t + 1 < SEQ) {
      const float* xpn =
          xproj + ((size_t)(t + 1) * BATCH + (threadIdx.x & 127)) * HID + colbase;
      __builtin_prefetch(xpn, 0, 3);               // -> global_prefetch_b8
    }

    const bf16* arow = hin + (size_t)(rowbase + l16) * HID;
    v8f acc0 = {};
    v8f acc1 = {};
#pragma unroll
    for (int kb = 0; kb < HID; kb += 32) {
      const v16bf a  = load_fragA_bf16(arow, kb, khalf);
      const v16bf w0 = load_fragB_bf16(b0row, kb, khalf);
      const v16bf w1 = load_fragB_bf16(b1row, kb, khalf);
      acc0 = __builtin_amdgcn_wmma_f32_16x16x32_bf16(false, a, false, w0, (short)0, acc0, false, false);
      acc1 = __builtin_amdgcn_wmma_f32_16x16x32_bf16(false, a, false, w1, (short)0, acc1, false, false);
    }

    const float* xp = xproj + (size_t)t * BATCH * HID;
    const int m0 = rowbase + khalf * 8;
#pragma unroll
    for (int r = 0; r < 8; ++r) {
      const int mb = m0 + r;
      const float v0 = tanhf(acc0[r] + xp[(size_t)mb * HID + colbase + l16]);
      const float v1 = tanhf(acc1[r] + xp[(size_t)mb * HID + colbase + 16 + l16]);
      hout[(size_t)mb * HID + colbase + l16]      = (bf16)v0;
      hout[(size_t)mb * HID + colbase + 16 + l16] = (bf16)v1;
    }

    grid_barrier(ctr, (unsigned)(RNN_NWG * (t + 1)), use_cluster);
  }
}

// ---------------- Kernel C: final FC  out = h_last @ W_fc^T + b_fc  (128x16) ----------------
__global__ void __launch_bounds__(256)
k_fc(const bf16* __restrict__ hlast, const float* __restrict__ W_fc,
     const float* __restrict__ b_fc, float* __restrict__ out)
{
  const int i = blockIdx.x * 256 + threadIdx.x;
  if (i >= BATCH * NCLASS) return;
  const int b = i >> 4;
  const int c = i & 15;
  const bf16*  h = hlast + (size_t)b * HID;
  const float* w = W_fc + (size_t)c * HID;
  float s = b_fc[c];
#pragma unroll 8
  for (int k = 0; k < HID; ++k) s += (float)h[k] * w[k];
  out[i] = s;
}

extern "C" void kernel_launch(void* const* d_in, const int* in_sizes, int n_in,
                              void* d_out, int out_size, void* d_ws, size_t ws_size,
                              hipStream_t stream)
{
  (void)in_sizes; (void)n_in; (void)out_size; (void)ws_size;
  const int*   x    = (const int*)  d_in[0];
  const float* emb  = (const float*)d_in[1];
  const float* W_ih = (const float*)d_in[2];
  const float* W_hh = (const float*)d_in[3];
  const float* b_ih = (const float*)d_in[4];
  const float* b_hh = (const float*)d_in[5];
  const float* W_fc = (const float*)d_in[6];
  const float* b_fc = (const float*)d_in[7];
  float* out = (float*)d_out;

  // Workspace layout (requires ~145 MiB):
  //   xproj   f32  [SEQ*BATCH, HID]   128 MiB
  //   emb_bf  bf16 [VOCAB, EMB]        15.6 MiB
  //   wih_bf  bf16 [HID, EMB]          256 KiB
  //   hb0/hb1 bf16 [BATCH, HID]        2 x 128 KiB
  //   ctr     u32 barrier counter
  char* ws = (char*)d_ws;
  const size_t XPROJ_BYTES = (size_t)MROWS * HID * sizeof(float);
  const size_t EMBB_BYTES  = (size_t)VOCAB * EMB * sizeof(bf16);
  const size_t WIHB_BYTES  = (size_t)HID * EMB * sizeof(bf16);
  const size_t HBUF_BYTES  = (size_t)BATCH * HID * sizeof(bf16);
  float*    xproj  = (float*)ws;
  bf16*     emb_bf = (bf16*)(ws + XPROJ_BYTES);
  bf16*     wih_bf = (bf16*)(ws + XPROJ_BYTES + EMBB_BYTES);
  bf16*     hb0    = (bf16*)(ws + XPROJ_BYTES + EMBB_BYTES + WIHB_BYTES);
  bf16*     hb1    = (bf16*)(ws + XPROJ_BYTES + EMBB_BYTES + WIHB_BYTES + HBUF_BYTES);
  unsigned* ctr    = (unsigned*)(ws + XPROJ_BYTES + EMBB_BYTES + WIHB_BYTES + 2 * HBUF_BYTES);

  // h0 = 0 and barrier counter = 0 (hb1 is fully overwritten at t=0 before any read).
  hipMemsetAsync(hb0, 0, 2 * HBUF_BYTES + 256, stream);

  k_prep<<<(NCH_EMB + NCH_WIH + 255) / 256, 256, 0, stream>>>(emb, W_ih, emb_bf, wih_bf);
  dim3 gridA(MROWS / 128, HID / 64);
  k_embed_xproj<<<gridA, 256, 0, stream>>>(x, emb_bf, wih_bf, b_ih, b_hh, xproj);
  k_rnn_scan<<<RNN_NWG, 256, 0, stream>>>(W_hh, xproj, hb0, hb1, ctr);
  k_fc<<<(BATCH * NCLASS + 255) / 256, 256, 0, stream>>>(hb0, W_fc, b_fc, out);
}
